// SlidingWindowAttention_21586505630196
// MI455X (gfx1250) — matrix operand order repaired and checked by reading, not compiled, not run on hardware
//
#include <hip/hip_runtime.h>
#include <cstdint>
#include <cstddef>

// Problem constants (from reference): B=8, S=4096, E=512, H=8, D=64, WIN=5
#define S_LEN  4096
#define EMB    512
#define NHEAD  8
#define HDIM   64
#define HALF_W 2

typedef __attribute__((ext_vector_type(16))) __bf16 bf16x16;
typedef __attribute__((ext_vector_type(8)))  float  f32x8;
typedef __attribute__((ext_vector_type(4)))  int    i32x4;

static __device__ __forceinline__ float bf2f(__bf16 b) {
  union { unsigned short s; __bf16 b; } in;
  in.b = b;
  union { unsigned u; float f; } out;
  out.u = ((unsigned)in.s) << 16;
  return out.f;
}

static __device__ __forceinline__ __bf16 f2bf(float f) {
  union { float f; unsigned u; } in;
  in.f = f;
  unsigned r = in.u + 0x7FFFu + ((in.u >> 16) & 1u);  // round-to-nearest-even
  union { unsigned short s; __bf16 b; } out;
  out.s = (unsigned short)(r >> 16);
  return out.b;
}

// ---------------------------------------------------------------------------
// Async global->LDS copy of 16 bytes (CDNA5 GLOBAL_LOAD_ASYNC_TO_LDS_B128,
// tracked by ASYNCcnt). Builtin signature (from hipcc diagnostic):
//   void(int4 AS1* gsrc, int4 AS3* ldst, imm int offset, imm int cpol)
// Fallback: sync load + ds_store.
// ---------------------------------------------------------------------------
#if __has_builtin(__builtin_amdgcn_global_load_async_to_lds_b128)
#define HAVE_ASYNC_LDS 1
static __device__ __forceinline__ void async_cp16(const __bf16* g, __bf16* l) {
  __builtin_amdgcn_global_load_async_to_lds_b128(
      (__attribute__((address_space(1))) i32x4*)g,
      (__attribute__((address_space(3))) i32x4*)l,
      /*offset=*/0, /*cpol=*/0);
}
#else
#define HAVE_ASYNC_LDS 0
static __device__ __forceinline__ void async_cp16(const __bf16* g, __bf16* l) {
  *(uint4*)l = *(const uint4*)g;
}
#endif

static __device__ __forceinline__ void wait_async_le4() {
#if HAVE_ASYNC_LDS
#if __has_builtin(__builtin_amdgcn_s_wait_asynccnt)
  __builtin_amdgcn_s_wait_asynccnt(4);
#else
  asm volatile("s_wait_asynccnt 0x4" ::: "memory");
#endif
#endif
}
static __device__ __forceinline__ void wait_async_0() {
#if HAVE_ASYNC_LDS
#if __has_builtin(__builtin_amdgcn_s_wait_asynccnt)
  __builtin_amdgcn_s_wait_asynccnt(0);
#else
  asm volatile("s_wait_asynccnt 0x0" ::: "memory");
#endif
#endif
}

// ---------------------------------------------------------------------------
// f32 -> bf16 conversion, 4 elements per thread (float4 in, 8B out)
// ---------------------------------------------------------------------------
__global__ __launch_bounds__(256) void cvt_f32_to_bf16(const float* __restrict__ in,
                                                       __bf16* __restrict__ out,
                                                       int n4) {
  int i = blockIdx.x * blockDim.x + threadIdx.x;
  if (i >= n4) return;
  const float4 f = ((const float4*)in)[i];
  union { __bf16 b[4]; uint2 u; } p;
  p.b[0] = f2bf(f.x); p.b[1] = f2bf(f.y); p.b[2] = f2bf(f.z); p.b[3] = f2bf(f.w);
  ((uint2*)out)[i] = p.u;
}

// ---------------------------------------------------------------------------
// Y[M,512] = X[M,512](bf16) @ W[512,512](bf16, row-major N,K)^T + bias
//
// Block: 256 threads = 8 wave32. Block tile 128(M) x 128(N), K stepped by 32.
// LDS double-buffered: sA[2] 128x32, sB[2] 128x32 bf16 (32 KB total).
// Copies use GLOBAL_LOAD_ASYNC_TO_LDS_B128 (4 per thread per K-step),
// double-buffered with s_wait_asynccnt: while computing slice k, slice k+1
// streams into the other buffer.
//
// Wave (mw 0..3, nw 0..1) computes a 32(M) x 64(N) strip: 2 A-frags x
// 4 B-frags = 8 f32 16x16 accumulators, 8 back-to-back WMMAs per K-step.
//
// Fragment layouts per CDNA5 ISA 7.12.2 (bf16):
//   A 16x32: lane = row (lane&15); lane-half selects which contiguous 8-run
//            of K in each 16-K group -> two 16B LDS reads per lane
//   B 32x16: lane = col (lane&15); lane-half selects the same K 8-runs as A
//            -> two 16B LDS reads per lane
//   C/D f32 16x16: VGPR r: M = r + 8*(lane>=16), N = lane&15
// ---------------------------------------------------------------------------
union Frag16 { bf16x16 v; uint4 q[2]; };

__global__ __launch_bounds__(256) void gemm_xwT_wmma(
    const __bf16* __restrict__ X,    // [M, 512]
    const __bf16* __restrict__ W,    // [512, 512]  (N, K) row-major
    const float*  __restrict__ bias, // [512]
    __bf16* __restrict__ outb,       // bf16 output (or null)
    float*  __restrict__ outf)       // f32 output (or null)
{
  __shared__ __bf16 sA[2][128 * 32];
  __shared__ __bf16 sB[2][128 * 32];

  const int tid  = threadIdx.x;
  const int lane = tid & 31;
  const int wave = tid >> 5;
  const int mw = wave & 3;          // 0..3 (M sub-tile)
  const int nw = wave >> 2;         // 0..1 (N sub-tile)
  const int mblk = blockIdx.x * 128;
  const int nblk = blockIdx.y * 128;
  const int msub = mw * 32;
  const int nsub = nw * 64;

  // copy assignment: thread copies 32B of one A row and 32B of one B row
  const int crow = tid >> 1;        // 0..127
  const int cseg = (tid & 1) * 16;  // bf16 offset within 32-element row slice

  const __bf16* gA = X + (size_t)(mblk + crow) * EMB + cseg;
  const __bf16* gB = W + (size_t)(nblk + crow) * EMB + cseg;
  __bf16* lA0 = &sA[0][crow * 32 + cseg];
  __bf16* lB0 = &sB[0][crow * 32 + cseg];
  __bf16* lA1 = &sA[1][crow * 32 + cseg];
  __bf16* lB1 = &sB[1][crow * 32 + cseg];

  f32x8 acc[2][4] = {};

  // prime buffer 0 with k0 = 0
  async_cp16(gA, lA0);     async_cp16(gA + 8, lA0 + 8);
  async_cp16(gB, lB0);     async_cp16(gB + 8, lB0 + 8);

  const int ahalf16 = (lane >> 4) * 16;  // byte offset selector (A runs, B runs)
  const int arow0 = msub + (lane & 15);

  for (int kc = 0; kc < 16; ++kc) {
    const int cur = kc & 1;
    if (kc < 15) {
      const int knxt = (kc + 1) * 32;
      __bf16* lA = cur ? lA0 : lA1;   // next buffer = cur ^ 1
      __bf16* lB = cur ? lB0 : lB1;
      async_cp16(gA + knxt, lA);      async_cp16(gA + knxt + 8, lA + 8);
      async_cp16(gB + knxt, lB);      async_cp16(gB + knxt + 8, lB + 8);
      wait_async_le4();               // older 4 (slice kc) have landed
    } else {
      wait_async_0();
    }
    __syncthreads();                  // whole tile visible to all waves

    // A fragments for this wave's two 16-row groups
    Frag16 a[2];
#pragma unroll
    for (int f = 0; f < 2; ++f) {
      const char* ab = (const char*)(&sA[cur][(arow0 + f * 16) * 32]);
      a[f].q[0] = *(const uint4*)(ab + ahalf16);        // K {0..7 | 8..15}
      a[f].q[1] = *(const uint4*)(ab + 32 + ahalf16);   // K 16 + {0..7 | 8..15}
    }
#pragma unroll
    for (int j = 0; j < 4; ++j) {
      const int c = nsub + j * 16 + (lane & 15);        // W row (N index)
      const char* bb = (const char*)(&sB[cur][c * 32]) + ahalf16; // K {0..7 | 8..15}, 16 + same
      Frag16 b;
      b.q[0] = *(const uint4*)(bb);
      b.q[1] = *(const uint4*)(bb + 32);
      acc[0][j] = __builtin_amdgcn_wmma_f32_16x16x32_bf16(
          false, a[0].v, false, b.v, (short)0, acc[0][j], false, false);
      acc[1][j] = __builtin_amdgcn_wmma_f32_16x16x32_bf16(
          false, a[1].v, false, b.v, (short)0, acc[1][j], false, false);
    }
    __syncthreads();                  // done reading cur before it is refilled
  }

#pragma unroll
  for (int f = 0; f < 2; ++f) {
#pragma unroll
    for (int j = 0; j < 4; ++j) {
      const int col = nblk + nsub + j * 16 + (lane & 15);
      const float bv = bias[col];
#pragma unroll
      for (int r = 0; r < 8; ++r) {
        const int row = mblk + msub + f * 16 + r + ((lane >> 4) << 3);
        const float val = acc[f][j][r] + bv;
        if (outb) outb[(size_t)row * EMB + col] = f2bf(val);
        else      outf[(size_t)row * EMB + col] = val;
      }
    }
  }
}

// ---------------------------------------------------------------------------
// Sliding-window attention (WIN=5, D=64): one thread per (b, s, h).
// q,k,v are bf16 [B*S, 512] with head h at columns h*64..h*64+63.
// Negligible FLOPs vs the GEMMs -> plain VALU f32 math.
// ---------------------------------------------------------------------------
__global__ __launch_bounds__(256) void swa_softmax(
    const __bf16* __restrict__ q, const __bf16* __restrict__ k,
    const __bf16* __restrict__ v, __bf16* __restrict__ ctx)
{
  const int g  = blockIdx.x * blockDim.x + threadIdx.x; // (b*S + s)*H + h
  const int h  = g & (NHEAD - 1);
  const int rs = g >> 3;                                // b*S + s
  const int s  = rs & (S_LEN - 1);
  const size_t qoff = (size_t)rs * EMB + h * HDIM;

  float qf[HDIM];
#pragma unroll
  for (int d = 0; d < HDIM; ++d) qf[d] = bf2f(q[qoff + d]);

  float sc[5];
  float mx = -3.4e38f;
#pragma unroll
  for (int w = 0; w < 5; ++w) {
    const int pos = s + w - HALF_W;
    float dot = -3.4e38f;
    if (pos >= 0 && pos < S_LEN) {
      const size_t koff = qoff + (size_t)(w - HALF_W) * EMB; // same batch
      float a = 0.f;
#pragma unroll
      for (int d = 0; d < HDIM; ++d) a += qf[d] * bf2f(k[koff + d]);
      dot = a * 0.125f;                                      // 1/sqrt(64)
    }
    sc[w] = dot;
    mx = fmaxf(mx, dot);
  }

  float p[5], denom = 0.f;
#pragma unroll
  for (int w = 0; w < 5; ++w) {
    p[w] = (sc[w] > -1e30f) ? __expf(sc[w] - mx) : 0.f;
    denom += p[w];
  }
  const float inv = 1.f / denom;

  float outv[HDIM];
#pragma unroll
  for (int d = 0; d < HDIM; ++d) outv[d] = 0.f;
#pragma unroll
  for (int w = 0; w < 5; ++w) {
    if (p[w] != 0.f) {
      const size_t voff = qoff + (size_t)(w - HALF_W) * EMB;
#pragma unroll
      for (int d = 0; d < HDIM; ++d) outv[d] += p[w] * bf2f(v[voff + d]);
    }
  }
#pragma unroll
  for (int d = 0; d < HDIM; ++d) ctx[qoff + d] = f2bf(outv[d] * inv);
}

// ---------------------------------------------------------------------------
// Host-side orchestration
// inputs: x, Wq, Wk, Wv, bq, bk, bv, Wo, bo
// ---------------------------------------------------------------------------
extern "C" void kernel_launch(void* const* d_in, const int* in_sizes, int n_in,
                              void* d_out, int out_size, void* d_ws, size_t ws_size,
                              hipStream_t stream) {
  const float* x  = (const float*)d_in[0];
  const float* Wq = (const float*)d_in[1];
  const float* Wk = (const float*)d_in[2];
  const float* Wv = (const float*)d_in[3];
  const float* bq = (const float*)d_in[4];
  const float* bk = (const float*)d_in[5];
  const float* bv = (const float*)d_in[6];
  const float* Wo = (const float*)d_in[7];
  const float* bo = (const float*)d_in[8];
  float* out = (float*)d_out;

  const int    M  = 8 * S_LEN;              // 32768 token rows
  const size_t xN = (size_t)M * EMB;        // 16,777,216
  const size_t wN = (size_t)EMB * EMB;      // 262,144

  char* ws = (char*)d_ws;
  size_t off = 0;
  __bf16* xb  = (__bf16*)(ws + off); off += xN * 2;
  __bf16* wqb = (__bf16*)(ws + off); off += wN * 2;
  __bf16* wkb = (__bf16*)(ws + off); off += wN * 2;
  __bf16* wvb = (__bf16*)(ws + off); off += wN * 2;
  __bf16* wob = (__bf16*)(ws + off); off += wN * 2;
  __bf16* qb  = (__bf16*)(ws + off); off += xN * 2;
  __bf16* kb  = (__bf16*)(ws + off); off += xN * 2;
  __bf16* vb  = (__bf16*)(ws + off); off += xN * 2;
  __bf16* cb  = (__bf16*)(ws + off); off += xN * 2;  // ~162 MB total

  // 1) f32 -> bf16 conversions (x once, 4 weight matrices)
  cvt_f32_to_bf16<<<(int)(xN / 4 / 256), 256, 0, stream>>>(x,  xb,  (int)(xN / 4));
  cvt_f32_to_bf16<<<(int)(wN / 4 / 256), 256, 0, stream>>>(Wq, wqb, (int)(wN / 4));
  cvt_f32_to_bf16<<<(int)(wN / 4 / 256), 256, 0, stream>>>(Wk, wkb, (int)(wN / 4));
  cvt_f32_to_bf16<<<(int)(wN / 4 / 256), 256, 0, stream>>>(Wv, wvb, (int)(wN / 4));
  cvt_f32_to_bf16<<<(int)(wN / 4 / 256), 256, 0, stream>>>(Wo, wob, (int)(wN / 4));

  // 2) Q/K/V projections: WMMA bf16 GEMMs, bf16 outputs
  dim3 ggrid(M / 128, EMB / 128);
  gemm_xwT_wmma<<<ggrid, 256, 0, stream>>>(xb, wqb, bq, qb, nullptr);
  gemm_xwT_wmma<<<ggrid, 256, 0, stream>>>(xb, wkb, bk, kb, nullptr);
  gemm_xwT_wmma<<<ggrid, 256, 0, stream>>>(xb, wvb, bv, vb, nullptr);

  // 3) Sliding-window softmax attention (VALU), bf16 context out
  swa_softmax<<<(M * NHEAD) / 256, 256, 0, stream>>>(qb, kb, vb, cb);

  // 4) Output projection: WMMA GEMM, f32 output to d_out
  gemm_xwT_wmma<<<ggrid, 256, 0, stream>>>(cb, wob, bo, nullptr, out);
}
